// EncodeBlock_83056077570430
// MI455X (gfx1250) — matrix operand order repaired and checked by reading, not compiled, not run on hardware
//
#include <hip/hip_runtime.h>
#include <stdint.h>

// ---------------------------------------------------------------------------
// Types for CDNA5 WMMA
// ---------------------------------------------------------------------------
typedef __attribute__((ext_vector_type(16))) __bf16 v16bf;
typedef __attribute__((ext_vector_type(8)))  float  v8f;

union FragU { uint32_t u[8]; v16bf v; };

__device__ __forceinline__ unsigned short f2bf(float f) {
    unsigned int u = __float_as_uint(f);
    unsigned int r = u + 0x7FFFu + ((u >> 16) & 1u);
    return (unsigned short)(r >> 16);
}

// Load 16 bf16 as a WMMA fragment from two 16B-aligned 16-byte chunks.
__device__ __forceinline__ v16bf ld_frag(const unsigned short* p0,
                                         const unsigned short* p1) {
    uint4 a = *(const uint4*)p0;
    uint4 b = *(const uint4*)p1;
    FragU f;
    f.u[0] = a.x; f.u[1] = a.y; f.u[2] = a.z; f.u[3] = a.w;
    f.u[4] = b.x; f.u[5] = b.y; f.u[6] = b.z; f.u[7] = b.w;
    return f.v;
}

__device__ __forceinline__ v8f wmma_bf16(v16bf a, v16bf b, v8f c) {
    return __builtin_amdgcn_wmma_f32_16x16x32_bf16(
        false, a, false, b, (short)0, c, false, false);
}

// ---------------------------------------------------------------------------
// CDNA5 async global->LDS copy (ASYNCcnt-tracked, no VGPR round trip)
// ---------------------------------------------------------------------------
__device__ __forceinline__ void async_b128(const unsigned short* lds_dst,
                                           const unsigned short* gsrc) {
    uint32_t dst = (uint32_t)(uintptr_t)(const void*)lds_dst;  // LDS byte offset
    unsigned long long src = (unsigned long long)(uintptr_t)gsrc;
    asm volatile("global_load_async_to_lds_b128 %0, %1, off"
                 :: "v"(dst), "v"(src) : "memory");
}

// Copy 32 bf16 (64B) global -> LDS via 4 async b128 transfers.
__device__ __forceinline__ void async_copy_row(unsigned short* lds_dst,
                                               const unsigned short* gsrc) {
#pragma unroll
    for (int j = 0; j < 4; ++j) async_b128(lds_dst + 8 * j, gsrc + 8 * j);
}

__device__ __forceinline__ void wait_async0() {
    asm volatile("s_wait_asynccnt 0x0" ::: "memory");
}

// ---------------------------------------------------------------------------
// Elementwise f32 -> bf16 cast
// ---------------------------------------------------------------------------
__global__ void cast_f32_bf16(const float* __restrict__ in,
                              unsigned short* __restrict__ out, int n) {
    int i = blockIdx.x * blockDim.x + threadIdx.x;
    if (i < n) out[i] = f2bf(in[i]);
}

// ---------------------------------------------------------------------------
// Transpose + cast: src (K x N) f32 row-major -> dst (N x K) bf16 row-major
// ---------------------------------------------------------------------------
__global__ __launch_bounds__(256)
void transpose_cast(const float* __restrict__ src,
                    unsigned short* __restrict__ dst, int K, int N) {
    __shared__ float tile[32][33];
    int nb = blockIdx.x * 32;
    int kb = blockIdx.y * 32;
    int tx = threadIdx.x;
    int ty = threadIdx.y;
#pragma unroll
    for (int i = 0; i < 32; i += 8)
        tile[ty + i][tx] = src[(size_t)(kb + ty + i) * N + nb + tx];
    __syncthreads();
#pragma unroll
    for (int i = 0; i < 32; i += 8)
        dst[(size_t)(nb + ty + i) * K + kb + tx] = f2bf(tile[tx][ty + i]);
}

// ---------------------------------------------------------------------------
// Per-head V transpose: V (b, s, h*64+d) bf16 -> VT ((b*16+h), d, s) bf16
// ---------------------------------------------------------------------------
__global__ __launch_bounds__(256)
void transpose_v(const unsigned short* __restrict__ V,
                 unsigned short* __restrict__ VT) {
    __shared__ unsigned short tile[32][40];
    const int s0 = blockIdx.x * 32;
    const int d0 = blockIdx.y * 32;
    const int bh = blockIdx.z;                 // b*16 + h
    const int b  = bh >> 4, h = bh & 15;
    const unsigned short* src = V + (size_t)b * 2048 * 1024 + (size_t)h * 64;
    unsigned short* dst = VT + (size_t)bh * 64 * 2048;
    const int tx = threadIdx.x, ty = threadIdx.y;   // (32, 8)
#pragma unroll
    for (int i = 0; i < 32; i += 8)
        tile[ty + i][tx] = src[(size_t)(s0 + ty + i) * 1024 + d0 + tx];
    __syncthreads();
#pragma unroll
    for (int i = 0; i < 32; i += 8)
        dst[(size_t)(d0 + ty + i) * 2048 + s0 + tx] = tile[tx][ty + i];
}

// ---------------------------------------------------------------------------
// GEMM: out = A (MxK bf16) @ Bt^T (Bt is NxK bf16) + bias, optional ReLU.
// 128x128 tile, BK=64, double-buffered LDS filled by async global->LDS.
// 8 waves, each wave 64x32 (4x2 accumulators), 16 WMMAs per K-tile per wave.
// ---------------------------------------------------------------------------
#define GS 72   // LDS row stride (elements): 144B rows, conflict-free b128

__global__ __launch_bounds__(256)
void gemm_bf16(const unsigned short* __restrict__ A,
               const unsigned short* __restrict__ Bt,
               const float* __restrict__ bias,
               float* __restrict__ outF,
               unsigned short* __restrict__ outB,
               int M, int N, int K, int relu) {
    __shared__ __align__(16) unsigned short As[2][128 * GS];
    __shared__ __align__(16) unsigned short Bs[2][128 * GS];

    const int bn = blockIdx.x * 128;
    const int bm = blockIdx.y * 128;
    const int t  = threadIdx.x;
    const int lane = t & 31;
    const int wid  = t >> 5;
    const int hh   = lane >> 4;
    const int l16  = lane & 15;
    const int wm   = (wid & 1) * 64;
    const int wn   = (wid >> 1) * 32;

    v8f acc[4][2];
#pragma unroll
    for (int m = 0; m < 4; ++m)
#pragma unroll
        for (int n = 0; n < 2; ++n) acc[m][n] = (v8f){0,0,0,0,0,0,0,0};

    const int srow = t >> 1;            // 0..127
    const int scol = (t & 1) * 32;      // 0 or 32
    const unsigned short* Ag = A  + (size_t)(bm + srow) * K + scol;
    const unsigned short* Bg = Bt + (size_t)(bn + srow) * K + scol;

    const int akb = hh ? 8 : 0;
    const int bkb = hh ? 8 : 0;

    auto stage = [&](int buf, int k0) {
        async_copy_row(&As[buf][srow * GS + scol], Ag + k0);
        async_copy_row(&Bs[buf][srow * GS + scol], Bg + k0);
    };

    stage(0, 0);
    int cur = 0;
    for (int k0 = 0; k0 < K; k0 += 64) {
        wait_async0();
        __syncthreads();
        if (k0 + 64 < K) stage(cur ^ 1, k0 + 64);

#pragma unroll
        for (int ks = 0; ks < 2; ++ks) {
            // Preload all fragments for this K-step, then dense WMMA burst.
            v16bf bfrag[2], afrag[4];
#pragma unroll
            for (int n = 0; n < 2; ++n) {
                const unsigned short* p =
                    &Bs[cur][(wn + n * 16 + l16) * GS + ks * 32 + bkb];
                bfrag[n] = ld_frag(p, p + 16);
            }
#pragma unroll
            for (int m = 0; m < 4; ++m) {
                const unsigned short* p =
                    &As[cur][(wm + m * 16 + l16) * GS + ks * 32 + akb];
                afrag[m] = ld_frag(p, p + 16);
            }
#pragma unroll
            for (int m = 0; m < 4; ++m)
#pragma unroll
                for (int n = 0; n < 2; ++n)
                    acc[m][n] = wmma_bf16(afrag[m], bfrag[n], acc[m][n]);
        }
        cur ^= 1;
    }

#pragma unroll
    for (int m = 0; m < 4; ++m)
#pragma unroll
        for (int n = 0; n < 2; ++n) {
            const int col = bn + wn + n * 16 + l16;
            const float bv = bias[col];
#pragma unroll
            for (int r = 0; r < 8; ++r) {
                const int row = bm + wm + m * 16 + r + hh * 8;
                float v = acc[m][n][r] + bv;
                if (relu) v = fmaxf(v, 0.0f);
                if (outF) outF[(size_t)row * N + col] = v;
                if (outB) outB[(size_t)row * N + col] = f2bf(v);
            }
        }
}

// ---------------------------------------------------------------------------
// Flash attention: one block per (64-query tile, head, batch), 4 waves.
// K and V^T tiles double-buffered in LDS via async copies.
// ---------------------------------------------------------------------------
#define SEQ   2048
#define DMOD  1024

__global__ __launch_bounds__(128)
void attention_fwd(const unsigned short* __restrict__ Qg,
                   const unsigned short* __restrict__ Kg,
                   const unsigned short* __restrict__ VTg,
                   unsigned short* __restrict__ Og) {
    const int q0 = blockIdx.x * 64;
    const int h  = blockIdx.y;
    const int b  = blockIdx.z;
    const size_t base = (size_t)b * SEQ * DMOD + (size_t)h * 64;
    const unsigned short* Qp  = Qg + base;
    const unsigned short* Kp  = Kg + base;
    const unsigned short* VTp = VTg + (size_t)(b * 16 + h) * 64 * SEQ;
    unsigned short*       Op  = Og + base;

    const int t    = threadIdx.x;
    const int lane = t & 31;
    const int wid  = t >> 5;
    const int hh   = lane >> 4;
    const int l16  = lane & 15;
    const int qrow = q0 + wid * 16;

    __shared__ __align__(16) unsigned short Ks[2][64 * GS];   // (kv, d)
    __shared__ __align__(16) unsigned short Vs[2][64 * GS];   // (d, kv)
    __shared__ __align__(16) unsigned short Ps[4][16 * GS];   // per-wave P

    // Q fragments (A layout), resident for the whole block
    v16bf aq[2];
    {
        const int row = qrow + l16;
        const int kb  = hh ? 8 : 0;
#pragma unroll
        for (int kk = 0; kk < 2; ++kk) {
            const unsigned short* p = Qp + (size_t)row * DMOD + kk * 32 + kb;
            aq[kk] = ld_frag(p, p + 16);
        }
    }

    float mrun[8], lsum[8];
#pragma unroll
    for (int r = 0; r < 8; ++r) { mrun[r] = -1e30f; lsum[r] = 0.0f; }
    v8f o[4];
#pragma unroll
    for (int d = 0; d < 4; ++d) o[d] = (v8f){0,0,0,0,0,0,0,0};

    const int srow = t >> 1;          // 0..63
    const int scol = (t & 1) * 32;    // 0 or 32
    const float sc = 0.125f;          // 1/sqrt(64)
    const int akb = hh ? 8 : 0;
    const int bkb = hh ? 8 : 0;

    auto stage_kv = [&](int buf, int kv0) {
        async_copy_row(&Ks[buf][srow * GS + scol],
                       Kp + (size_t)(kv0 + srow) * DMOD + scol);
        async_copy_row(&Vs[buf][srow * GS + scol],
                       VTp + (size_t)srow * SEQ + kv0 + scol);
    };

    stage_kv(0, 0);
    int cur = 0;
    for (int kv0 = 0; kv0 < SEQ; kv0 += 64) {
        wait_async0();
        __syncthreads();
        if (kv0 + 64 < SEQ) stage_kv(cur ^ 1, kv0 + 64);

        // Scores: 16x64 strip per wave; 4 independent WMMAs per K-step
        v8f s[4];
#pragma unroll
        for (int n = 0; n < 4; ++n) s[n] = (v8f){0,0,0,0,0,0,0,0};
#pragma unroll
        for (int kk = 0; kk < 2; ++kk) {
            v16bf bk[4];
#pragma unroll
            for (int n = 0; n < 4; ++n) {
                const unsigned short* p =
                    &Ks[cur][(n * 16 + l16) * GS + kk * 32 + bkb];
                bk[n] = ld_frag(p, p + 16);
            }
#pragma unroll
            for (int n = 0; n < 4; ++n)
                s[n] = wmma_bf16(aq[kk], bk[n], s[n]);
        }

        // Online softmax (rows live across 16 lanes of a half-wave)
        float alpha[8];
#pragma unroll
        for (int r = 0; r < 8; ++r) {
            float mx = fmaxf(fmaxf(s[0][r], s[1][r]), fmaxf(s[2][r], s[3][r]));
#pragma unroll
            for (int sh = 8; sh >= 1; sh >>= 1)
                mx = fmaxf(mx, __shfl_xor(mx, sh));
            float mnew = fmaxf(mrun[r], mx * sc);
            alpha[r] = __expf(mrun[r] - mnew);
            mrun[r] = mnew;
        }
        float rsum[8];
#pragma unroll
        for (int r = 0; r < 8; ++r) rsum[r] = 0.0f;
#pragma unroll
        for (int n = 0; n < 4; ++n)
#pragma unroll
            for (int r = 0; r < 8; ++r) {
                float pv = __expf(s[n][r] * sc - mrun[r]);
                s[n][r] = pv;
                rsum[r] += pv;
            }
#pragma unroll
        for (int r = 0; r < 8; ++r) {
#pragma unroll
            for (int sh = 8; sh >= 1; sh >>= 1)
                rsum[r] += __shfl_xor(rsum[r], sh);
            lsum[r] = lsum[r] * alpha[r] + rsum[r];
        }
#pragma unroll
        for (int d = 0; d < 4; ++d)
#pragma unroll
            for (int r = 0; r < 8; ++r) o[d][r] *= alpha[r];

        // Write P strip (C layout -> LDS), reload as A fragments
        unsigned short* pw = &Ps[wid][0];
#pragma unroll
        for (int n = 0; n < 4; ++n)
#pragma unroll
            for (int r = 0; r < 8; ++r)
                pw[(r + hh * 8) * GS + n * 16 + l16] = f2bf(s[n][r]);
        __syncthreads();

        // o += P @ V ; 4 independent WMMAs per K-step
#pragma unroll
        for (int kk = 0; kk < 2; ++kk) {
            const unsigned short* pp = &Ps[wid][l16 * GS + kk * 32 + akb];
            v16bf ap = ld_frag(pp, pp + 16);
            v16bf bv[4];
#pragma unroll
            for (int d = 0; d < 4; ++d) {
                const unsigned short* vp =
                    &Vs[cur][(d * 16 + l16) * GS + kk * 32 + bkb];
                bv[d] = ld_frag(vp, vp + 16);
            }
#pragma unroll
            for (int d = 0; d < 4; ++d)
                o[d] = wmma_bf16(ap, bv[d], o[d]);
        }
        cur ^= 1;
    }

    float inv[8];
#pragma unroll
    for (int r = 0; r < 8; ++r) inv[r] = 1.0f / lsum[r];
#pragma unroll
    for (int d = 0; d < 4; ++d)
#pragma unroll
        for (int r = 0; r < 8; ++r) {
            const int row = qrow + r + hh * 8;
            const int col = d * 16 + l16;
            Op[(size_t)row * DMOD + col] = f2bf(o[d][r] * inv[r]);
        }
}

// ---------------------------------------------------------------------------
// Fused residual + LayerNorm over D=1024. One block per row, 256 threads.
// ---------------------------------------------------------------------------
__global__ __launch_bounds__(256)
void ln_residual(const float* __restrict__ A, const float* __restrict__ B,
                 const float* __restrict__ G, const float* __restrict__ Bb,
                 float* __restrict__ outF, unsigned short* __restrict__ outB) {
    const int D = 1024;
    const size_t off = (size_t)blockIdx.x * D;
    float v[4];
    float s = 0.0f, s2 = 0.0f;
#pragma unroll
    for (int i = 0; i < 4; ++i) {
        int idx = threadIdx.x + i * 256;
        float x = A[off + idx] + B[off + idx];
        v[i] = x; s += x; s2 += x * x;
    }
#pragma unroll
    for (int m = 16; m >= 1; m >>= 1) {
        s  += __shfl_xor(s, m);
        s2 += __shfl_xor(s2, m);
    }
    __shared__ float rs[8], rs2[8];
    const int wid = threadIdx.x >> 5, lane = threadIdx.x & 31;
    if (lane == 0) { rs[wid] = s; rs2[wid] = s2; }
    __syncthreads();
    if (wid == 0) {
        float a  = (lane < 8) ? rs[lane]  : 0.0f;
        float a2 = (lane < 8) ? rs2[lane] : 0.0f;
#pragma unroll
        for (int m = 4; m >= 1; m >>= 1) {
            a  += __shfl_xor(a, m);
            a2 += __shfl_xor(a2, m);
        }
        if (lane == 0) { rs[0] = a; rs2[0] = a2; }
    }
    __syncthreads();
    const float mu   = rs[0] * (1.0f / 1024.0f);
    const float var  = rs2[0] * (1.0f / 1024.0f) - mu * mu;
    const float rstd = rsqrtf(var + 1e-5f);
#pragma unroll
    for (int i = 0; i < 4; ++i) {
        int idx = threadIdx.x + i * 256;
        float y = (v[i] - mu) * rstd * G[idx] + Bb[idx];
        outF[off + idx] = y;
        if (outB) outB[off + idx] = f2bf(y);
    }
}

// ---------------------------------------------------------------------------
// Host orchestration
// ---------------------------------------------------------------------------
extern "C" void kernel_launch(void* const* d_in, const int* in_sizes, int n_in,
                              void* d_out, int out_size, void* d_ws, size_t ws_size,
                              hipStream_t stream) {
    (void)in_sizes; (void)n_in; (void)out_size; (void)ws_size;
    const int BATCH = 4, S = SEQ, D = DMOD, DFF = 4096;
    const size_t ROWS = (size_t)BATCH * S;          // 8192

    const float* x    = (const float*)d_in[0];
    const float* Wq   = (const float*)d_in[1];
    const float* bq   = (const float*)d_in[2];
    const float* Wk   = (const float*)d_in[3];
    const float* bk   = (const float*)d_in[4];
    const float* Wv   = (const float*)d_in[5];
    const float* bv   = (const float*)d_in[6];
    const float* Wo   = (const float*)d_in[7];
    const float* bo   = (const float*)d_in[8];
    const float* ln1g = (const float*)d_in[9];
    const float* ln1b = (const float*)d_in[10];
    const float* W1   = (const float*)d_in[11];
    const float* b1   = (const float*)d_in[12];
    const float* W2   = (const float*)d_in[13];
    const float* b2   = (const float*)d_in[14];
    const float* ln2g = (const float*)d_in[15];
    const float* ln2b = (const float*)d_in[16];
    float* out = (float*)d_out;

    char* ws = (char*)d_ws;
    size_t off = 0;
    auto take = [&](size_t bytes) {
        size_t o = off; off += (bytes + 255) & ~(size_t)255; return o;
    };
    unsigned short* xb   = (unsigned short*)(ws + take(ROWS * D * 2));
    unsigned short* WqT  = (unsigned short*)(ws + take((size_t)D * D * 2));
    unsigned short* WkT  = (unsigned short*)(ws + take((size_t)D * D * 2));
    unsigned short* WvT  = (unsigned short*)(ws + take((size_t)D * D * 2));
    unsigned short* WoT  = (unsigned short*)(ws + take((size_t)D * D * 2));
    unsigned short* W1T  = (unsigned short*)(ws + take((size_t)D * DFF * 2));
    unsigned short* W2T  = (unsigned short*)(ws + take((size_t)DFF * D * 2));
    size_t qoff = take(ROWS * D * 2);
    unsigned short* Qb   = (unsigned short*)(ws + qoff);
    unsigned short* Kb   = (unsigned short*)(ws + take(ROWS * D * 2));
    unsigned short* Vb   = (unsigned short*)(ws + take(ROWS * D * 2));
    unsigned short* VTb  = (unsigned short*)(ws + take(ROWS * D * 2));
    unsigned short* Ab   = (unsigned short*)(ws + take(ROWS * D * 2));
    float*          oF   = (float*)(ws + take(ROWS * D * 4));
    float*          hF   = (float*)(ws + take(ROWS * D * 4));
    unsigned short* hB   = (unsigned short*)(ws + take(ROWS * D * 2));
    // FFN1 output (ROWS x DFF bf16 = 64MB) reuses dead Q/K/V/VT region
    unsigned short* f1b  = (unsigned short*)(ws + qoff);
    // FFN2 output reuses the dead O-proj buffer
    float*          f2F  = oF;

    // 1) Casts / weight transposes
    {
        int n = (int)(ROWS * D);
        cast_f32_bf16<<<(n + 255) / 256, 256, 0, stream>>>(x, xb, n);
        dim3 blk(32, 8);
        transpose_cast<<<dim3(D / 32, D / 32), blk, 0, stream>>>(Wq, WqT, D, D);
        transpose_cast<<<dim3(D / 32, D / 32), blk, 0, stream>>>(Wk, WkT, D, D);
        transpose_cast<<<dim3(D / 32, D / 32), blk, 0, stream>>>(Wv, WvT, D, D);
        transpose_cast<<<dim3(D / 32, D / 32), blk, 0, stream>>>(Wo, WoT, D, D);
        transpose_cast<<<dim3(DFF / 32, D / 32), blk, 0, stream>>>(W1, W1T, D, DFF);
        transpose_cast<<<dim3(D / 32, DFF / 32), blk, 0, stream>>>(W2, W2T, DFF, D);
    }

    // 2) QKV projections (bf16 outputs)
    {
        dim3 g(D / 128, (unsigned)(ROWS / 128)), blk(256);
        gemm_bf16<<<g, blk, 0, stream>>>(xb, WqT, bq, nullptr, Qb,
                                         (int)ROWS, D, D, 0);
        gemm_bf16<<<g, blk, 0, stream>>>(xb, WkT, bk, nullptr, Kb,
                                         (int)ROWS, D, D, 0);
        gemm_bf16<<<g, blk, 0, stream>>>(xb, WvT, bv, nullptr, Vb,
                                         (int)ROWS, D, D, 0);
    }

    // 3) Per-head V transpose for the P@V WMMA path
    transpose_v<<<dim3(S / 32, 2, BATCH * 16), dim3(32, 8), 0, stream>>>(Vb, VTb);

    // 4) Attention
    attention_fwd<<<dim3(S / 64, 16, BATCH), 128, 0, stream>>>(Qb, Kb, VTb, Ab);

    // 5) Output projection (f32)
    gemm_bf16<<<dim3(D / 128, (unsigned)(ROWS / 128)), 256, 0, stream>>>(
        Ab, WoT, bo, oF, nullptr, (int)ROWS, D, D, 0);

    // 6) h = LN(attn_out + x)
    ln_residual<<<(unsigned)ROWS, 256, 0, stream>>>(oF, x, ln1g, ln1b, hF, hB);

    // 7) FFN1: relu(h @ W1 + b1) -> bf16
    gemm_bf16<<<dim3(DFF / 128, (unsigned)(ROWS / 128)), 256, 0, stream>>>(
        hB, W1T, b1, nullptr, f1b, (int)ROWS, DFF, D, 1);

    // 8) FFN2: f1 @ W2 + b2 -> f32
    gemm_bf16<<<dim3(D / 128, (unsigned)(ROWS / 128)), 256, 0, stream>>>(
        f1b, W2T, b2, f2F, nullptr, (int)ROWS, D, DFF, 0);

    // 9) out = LN(f2 + h)
    ln_residual<<<(unsigned)ROWS, 256, 0, stream>>>(f2F, hF, ln2g, ln2b,
                                                    out, nullptr);
}
